// Block_73967926771745
// MI455X (gfx1250) — matrix operand order repaired and checked by reading, not compiled, not run on hardware
//
#include <hip/hip_runtime.h>
#include <math.h>

// ---------------------------------------------------------------------------
// MI455X (gfx1250) implementation, round 3.
// Compute-bound (~232 GFLOP, working set ~200MB fits the 192MB L2) -> all
// GEMM/conv via v_wmma_f32_16x16x32_bf16 (bf16 in, f32 acc).
// Round-3 change: 4x1 register blocking (M=64 x N=16 per wave) in the GEMM
// and implicit-GEMM conv kernels: the shared B fragment is loaded once per
// k-step and reused by 4 WMMAs (arith intensity 8 -> ~13 flops/byte of L2).
// All fragment gathers remain unconditional contiguous b128 loads:
//  * every K is a multiple of 32 (P rows padded 784->800, V transposed+padded)
//  * B operands stored N-major (transposed weights, NHWC activations, Vt)
//  * conv zero-padding via dword AND mask on safely-clamped loads
// ---------------------------------------------------------------------------

#define DEVINL __device__ __forceinline__
typedef unsigned short u16;
typedef unsigned int   u32;

typedef __attribute__((ext_vector_type(16))) __bf16 v16bf;
typedef __attribute__((ext_vector_type(8)))  float  v8f;

union BF16Frag { u16 u[16]; u32 w[8]; v16bf v; };

DEVINL u16 f2bf(float x) {
  u32 u = __float_as_uint(x);
  return (u16)((u + 0x7FFFu + ((u >> 16) & 1u)) >> 16);
}
DEVINL float gelu_exact(float x) {
  return 0.5f * x * (1.0f + erff(x * 0.70710678118654752440f));
}

// A fragment, row-major A[m][k] (lda even, k0 multiple of 32, base 4B aligned)
DEVINL v16bf frag_a_rm(const u16* __restrict__ A, int lda, int m0, int k0,
                       int lane) {
  const u32* r = (const u32*)(A + (size_t)(m0 + (lane & 15)) * lda + k0 +
                              ((lane >> 4) * 16));
  BF16Frag f;
  f.w[0] = r[0]; f.w[1] = r[1]; f.w[2] = r[2];  f.w[3] = r[3];
  f.w[4] = r[4]; f.w[5] = r[5]; f.w[6] = r[6]; f.w[7] = r[7];
  return f.v;
}
// B fragment from N-major storage Bt[n][k]: B[k][n] = Bt[n*ldbt + k]
DEVINL v16bf frag_b_nm(const u16* __restrict__ Bt, int ldbt, int k0, int n0,
                       int lane) {
  const u32* r = (const u32*)(Bt + (size_t)(n0 + (lane & 15)) * ldbt + k0 +
                              ((lane >> 4) * 16));
  BF16Frag f;
#pragma unroll
  for (int i = 0; i < 8; ++i) f.w[i] = r[i];
  return f.v;
}
DEVINL v8f wmma_bf16(v16bf a, v16bf b, v8f c) {
  return __builtin_amdgcn_wmma_f32_16x16x32_bf16(false, a, false, b,
                                                 (short)0, c, false, false);
}

// ---------------------------------------------------------------------------
__global__ void k_pack_bf16(const float* __restrict__ src,
                            u16* __restrict__ dst, int n) {
  int i = blockIdx.x * 256 + threadIdx.x;
  if (i < n) dst[i] = f2bf(src[i]);
}
// W[K][N] f32 -> Wt[N][K] bf16  (for GEMM B operands)
__global__ void k_pack_bf16_t(const float* __restrict__ src,
                              u16* __restrict__ dst, int K, int N) {
  int i = blockIdx.x * 256 + threadIdx.x;
  if (i >= K * N) return;
  int k = i / N, n = i % N;
  dst[(size_t)n * K + k] = f2bf(src[i]);
}
// conv W[Co][Ci][KHW] f32 -> [tap][Co][Ci] bf16
__global__ void k_pack_convw(const float* __restrict__ src,
                             u16* __restrict__ dst, int Co, int Ci, int KHW) {
  int i = blockIdx.x * 256 + threadIdx.x;
  if (i >= Co * Ci * KHW) return;
  int co = i / (Ci * KHW);
  int rem = i % (Ci * KHW);
  int ci = rem / KHW, t = rem % KHW;
  dst[((size_t)t * Co + co) * Ci + ci] = f2bf(src[i]);
}

// LayerNorm over C=256, one wave per token. Outputs token-major (== NHWC).
__global__ void k_layernorm(const float* __restrict__ x,
                            const float* __restrict__ g,
                            const float* __restrict__ bb,
                            u16* __restrict__ out_bf,
                            float* __restrict__ out_f) {
  int wave = blockIdx.x * 8 + (threadIdx.x >> 5);
  if (wave >= 3136) return;
  int lane = threadIdx.x & 31;
  const float* row = x + (size_t)wave * 256;
  float vals[8], s = 0.f, s2 = 0.f;
#pragma unroll
  for (int t = 0; t < 8; ++t) {
    float v = row[lane + 32 * t];
    vals[t] = v; s += v; s2 += v * v;
  }
#pragma unroll
  for (int o = 16; o >= 1; o >>= 1) {
    s  += __shfl_xor(s,  o, 32);
    s2 += __shfl_xor(s2, o, 32);
  }
  float mu  = s * (1.0f / 256.0f);
  float var = s2 * (1.0f / 256.0f) - mu * mu;
  float inv = rsqrtf(var + 1e-5f);
#pragma unroll
  for (int t = 0; t < 8; ++t) {
    int c = lane + 32 * t;
    float y = (vals[t] - mu) * inv * g[c] + bb[c];
    if (out_bf) out_bf[(size_t)wave * 256 + c] = f2bf(y);
    if (out_f)  out_f[(size_t)wave * 256 + c] = y;
  }
}

// GEMM: out[M,N] = A[M,K](bf16) * Bt[N,K]^T(bf16)  (+bias per col, +resid)
// 4x1 blocking: wave computes 64 rows x 16 cols; B fragment reused 4x.
// Requires M % 64 == 0, Ncols % 16 == 0, K % 32 == 0.
__global__ void k_gemm_bf16(const u16* __restrict__ A, int lda,
                            const u16* __restrict__ Bt, int ldbt,
                            int M, int Ncols, int K,
                            const float* __restrict__ bias,
                            const float* __restrict__ resid,
                            float* __restrict__ outf,
                            u16* __restrict__ outbf, int ldc) {
  int wave  = blockIdx.x * (blockDim.x >> 5) + (threadIdx.x >> 5);
  int tN    = Ncols >> 4;
  int total = (M >> 6) * tN;
  if (wave >= total) return;
  int lane = threadIdx.x & 31;
  int n0 = (wave % tN) << 4;
  int m0 = (wave / tN) << 6;
  v8f acc0 = {}, acc1 = {}, acc2 = {}, acc3 = {};
  for (int k0 = 0; k0 < K; k0 += 32) {
    v16bf b  = frag_b_nm(Bt, ldbt, k0, n0, lane);
    v16bf a0 = frag_a_rm(A, lda, m0,      k0, lane);
    v16bf a1 = frag_a_rm(A, lda, m0 + 16, k0, lane);
    v16bf a2 = frag_a_rm(A, lda, m0 + 32, k0, lane);
    v16bf a3 = frag_a_rm(A, lda, m0 + 48, k0, lane);
    acc0 = wmma_bf16(a0, b, acc0);
    acc1 = wmma_bf16(a1, b, acc1);
    acc2 = wmma_bf16(a2, b, acc2);
    acc3 = wmma_bf16(a3, b, acc3);
  }
  int n = n0 + (lane & 15);
  float bv = bias ? bias[n] : 0.f;
#pragma unroll
  for (int t = 0; t < 4; ++t) {
    v8f acc = (t == 0) ? acc0 : (t == 1) ? acc1 : (t == 2) ? acc2 : acc3;
    int mb = m0 + 16 * t + ((lane >> 4) << 3);
#pragma unroll
    for (int r = 0; r < 8; ++r) {
      float v = acc[r] + bv;
      size_t oi = (size_t)(mb + r) * ldc + n;
      if (resid) v += resid[oi];
      if (outf)  outf[oi]  = v;
      if (outbf) outbf[oi] = f2bf(v);
    }
  }
}

// Scores: S[b,h,i,j] = (q.k)*HD^-0.5 + rel[h,i,j]   (K=32, one WMMA/tile)
__global__ void k_attn_score(const u16* __restrict__ Q,
                             const u16* __restrict__ Km,
                             const float* __restrict__ rel,
                             float* __restrict__ S) {
  int wave  = blockIdx.x * 8 + (threadIdx.x >> 5);
  int total = 4 * 8 * 49 * 49;
  if (wave >= total) return;
  int lane = threadIdx.x & 31;
  int t  = wave;
  int j0 = (t % 49) << 4; t /= 49;
  int i0 = (t % 49) << 4; t /= 49;
  int h  = t & 7;  int b = t >> 3;
  const u16* qb = Q + (size_t)b * 784 * 256 + h * 32;
  const u16* kb = Km + (size_t)b * 784 * 256 + h * 32;
  v16bf a  = frag_a_rm(qb, 256, i0, 0, lane);
  v16bf bt = frag_b_nm(kb, 256, 0, j0, lane);
  v8f acc = {};
  acc = wmma_bf16(a, bt, acc);
  int n  = j0 + (lane & 15);
  int mb = i0 + ((lane >> 4) << 3);
#pragma unroll
  for (int r = 0; r < 8; ++r) {
    int i = mb + r;
    S[((size_t)(b * 8 + h) * 784 + i) * 784 + n] =
        acc[r] * 0.17677669529663689f + rel[((size_t)h * 784 + i) * 784 + n];
  }
}

// Row softmax over 784; writes bf16 probabilities into rows of stride 800,
// zero-filling the 16-column pad so the PV GEMM can run K=800 unmasked.
__global__ void k_softmax(const float* __restrict__ S, u16* __restrict__ P) {
  __shared__ float red[256];
  size_t row = blockIdx.x;
  const float* s = S + row * 784;
  int tid = threadIdx.x;
  float vals[4];
  float mx = -1e30f;
#pragma unroll
  for (int t = 0; t < 4; ++t) {
    int j = tid + 256 * t;
    float v = (j < 784) ? s[j] : -1e30f;
    vals[t] = v; mx = fmaxf(mx, v);
  }
  red[tid] = mx; __syncthreads();
  for (int o = 128; o > 0; o >>= 1) {
    if (tid < o) red[tid] = fmaxf(red[tid], red[tid + o]);
    __syncthreads();
  }
  mx = red[0]; __syncthreads();
  float sum = 0.f;
#pragma unroll
  for (int t = 0; t < 4; ++t) {
    int j = tid + 256 * t;
    if (j < 784) { vals[t] = expf(vals[t] - mx); sum += vals[t]; }
  }
  red[tid] = sum; __syncthreads();
  for (int o = 128; o > 0; o >>= 1) {
    if (tid < o) red[tid] += red[tid + o];
    __syncthreads();
  }
  float inv = 1.0f / red[0];
#pragma unroll
  for (int t = 0; t < 4; ++t) {
    int j = tid + 256 * t;
    if (j < 784)      P[row * 800 + j] = f2bf(vals[t] * inv);
    else if (j < 800) P[row * 800 + j] = 0;
  }
}

// V tokens [b*784+j][256] -> Vt [b*256+c][800] (zero pad j=784..799)
__global__ void k_vt(const u16* __restrict__ vb, u16* __restrict__ vt) {
  int i = blockIdx.x * 256 + threadIdx.x;
  if (i >= 4 * 256 * 800) return;
  int j = i % 800;
  int row = i / 800;
  int b = row >> 8, c = row & 255;
  vt[i] = (j < 784) ? vb[((size_t)(b * 784 + j)) * 256 + c] : (u16)0;
}

// O = P @ V  (M=784, N=32, K=800 padded) per (b,h); bf16 tokens out.
__global__ void k_attn_pv(const u16* __restrict__ P,
                          const u16* __restrict__ Vt,
                          u16* __restrict__ Ob) {
  int wave  = blockIdx.x * 8 + (threadIdx.x >> 5);
  int total = 4 * 8 * 49 * 2;
  if (wave >= total) return;
  int lane = threadIdx.x & 31;
  int t  = wave;
  int d0 = (t & 1) << 4; t >>= 1;
  int i0 = (t % 49) << 4; t /= 49;
  int h  = t & 7; int b = t >> 3;
  const u16* A  = P + (size_t)(b * 8 + h) * 784 * 800;
  const u16* Bt = Vt + (size_t)(b * 256 + h * 32) * 800;
  v8f acc = {};
  for (int k0 = 0; k0 < 800; k0 += 32) {
    v16bf a  = frag_a_rm(A, 800, i0, k0, lane);
    v16bf bb = frag_b_nm(Bt, 800, k0, d0, lane);
    acc = wmma_bf16(a, bb, acc);
  }
  u16* O = Ob + (size_t)b * 784 * 256 + h * 32;
  int n  = d0 + (lane & 15);
  int mb = i0 + ((lane >> 4) << 3);
#pragma unroll
  for (int r = 0; r < 8; ++r) O[(size_t)(mb + r) * 256 + n] = f2bf(acc[r]);
}

// Implicit-GEMM conv, NHWC bf16 activations [b][28*28][Ci],
// weights packed [tap][Co][Ci]. 4x1 blocking: wave = 64 Cout x 16 pixels,
// activation B fragment loaded once per k-step, reused by 4 WMMAs.
// Fused (+bias)->(gelu)->(BN) epilogue; NHWC f32/bf16 outputs (contiguous).
// Requires Co % 64 == 0, Ci % 32 == 0.
__global__ void k_conv_wmma(const u16* __restrict__ in, int Ci,
                            const u16* __restrict__ wpk,
                            int KH, int KW, int pad, int dil, int Co,
                            const float* __restrict__ bias, int do_gelu,
                            const float* __restrict__ bng,
                            const float* __restrict__ bnb,
                            const float* __restrict__ bnm,
                            const float* __restrict__ bnv,
                            float* __restrict__ outf,
                            u16* __restrict__ outbf,
                            int Co_ld, int co_off) {
  int wave  = blockIdx.x * (blockDim.x >> 5) + (threadIdx.x >> 5);
  int tCo   = Co >> 6;
  int total = 4 * tCo * 49;
  if (wave >= total) return;
  int lane = threadIdx.x & 31;
  int t   = wave;
  int px0 = (t % 49) << 4; t /= 49;
  int co0 = (t % tCo) << 6; t /= tCo;
  int b = t;

  int p   = px0 + (lane & 15);
  int py  = p / 28, pxx = p % 28;
  int bkb = (lane >> 4) * 16;
  const u16* inb = in + (size_t)b * 784 * Ci;

  v8f acc0 = {}, acc1 = {}, acc2 = {}, acc3 = {};
  for (int ky = 0; ky < KH; ++ky) {
    int iy = py - pad + ky * dil;
    bool yok = (iy >= 0) & (iy < 28);
    for (int kx = 0; kx < KW; ++kx) {
      int ix = pxx - pad + kx * dil;
      bool ok = yok & (ix >= 0) & (ix < 28);
      int pixoff = ok ? (iy * 28 + ix) : 0;           // clamped, always valid
      u32 msk = ok ? 0xFFFFFFFFu : 0u;                // zero via AND, no branch
      const u16* brow = inb + (size_t)pixoff * Ci + bkb;
      const u16* wt = wpk + (size_t)(ky * KW + kx) * Co * Ci;
      __builtin_prefetch((const void*)(wt + (size_t)co0 * Ci), 0, 3);
      for (int ci0 = 0; ci0 < Ci; ci0 += 32) {
        const u32* r = (const u32*)(brow + ci0);
        BF16Frag fb;
#pragma unroll
        for (int i = 0; i < 8; ++i) fb.w[i] = r[i] & msk;
        v16bf a0 = frag_a_rm(wt, Ci, co0,      ci0, lane);
        v16bf a1 = frag_a_rm(wt, Ci, co0 + 16, ci0, lane);
        v16bf a2 = frag_a_rm(wt, Ci, co0 + 32, ci0, lane);
        v16bf a3 = frag_a_rm(wt, Ci, co0 + 48, ci0, lane);
        acc0 = wmma_bf16(a0, fb.v, acc0);
        acc1 = wmma_bf16(a1, fb.v, acc1);
        acc2 = wmma_bf16(a2, fb.v, acc2);
        acc3 = wmma_bf16(a3, fb.v, acc3);
      }
    }
  }
  int n = px0 + (lane & 15);
#pragma unroll
  for (int t4 = 0; t4 < 4; ++t4) {
    v8f acc = (t4 == 0) ? acc0 : (t4 == 1) ? acc1 : (t4 == 2) ? acc2 : acc3;
    int mb = co0 + 16 * t4 + ((lane >> 4) << 3);
    size_t base = (size_t)(b * 784 + n) * Co_ld + co_off + mb;
#pragma unroll
    for (int r = 0; r < 8; ++r) {
      int co = mb + r;
      float v = acc[r];
      if (bias) v += bias[co];
      if (do_gelu) v = gelu_exact(v);
      if (bng) {
        float sc = bng[co] * rsqrtf(bnv[co] + 1e-5f);
        v = v * sc + (bnb[co] - bnm[co] * sc);
      }
      if (outf)  outf[base + r]  = v;
      if (outbf) outbf[base + r] = f2bf(v);
    }
  }
}

// channel mean/max over C=256 (token-major input) -> pool[b][2][784]
__global__ void k_sa_pool(const float* __restrict__ f, float* __restrict__ pool) {
  int i = blockIdx.x * 256 + threadIdx.x;
  if (i >= 4 * 784) return;
  int b = i / 784, p = i % 784;
  const float* row = f + (size_t)i * 256;
  float s = 0.f, mx = -1e30f;
  for (int c = 0; c < 256; ++c) {
    float v = row[c];
    s += v; mx = fmaxf(mx, v);
  }
  pool[((size_t)b * 2 + 0) * 784 + p] = s * (1.0f / 256.0f);
  pool[((size_t)b * 2 + 1) * 784 + p] = mx;
}

// 7x7 conv (2ch -> 1, pad 3) + sigmoid -> sa[b*784+p]
__global__ void k_sa_conv(const float* __restrict__ pool,
                          const float* __restrict__ w, float* __restrict__ sa) {
  int i = blockIdx.x * 256 + threadIdx.x;
  if (i >= 4 * 784) return;
  int b = i / 784, p = i % 784;
  int y = p / 28, x = p % 28;
  float acc = 0.f;
  for (int ch = 0; ch < 2; ++ch)
    for (int dy = 0; dy < 7; ++dy) {
      int iy = y - 3 + dy;
      if (iy < 0 || iy >= 28) continue;
      for (int dx = 0; dx < 7; ++dx) {
        int ix = x - 3 + dx;
        if (ix < 0 || ix >= 28) continue;
        acc += w[(ch * 7 + dy) * 7 + dx] *
               pool[((size_t)b * 2 + ch) * 784 + iy * 28 + ix];
      }
    }
  sa[i] = 1.0f / (1.0f + expf(-acc));
}

// fg = bf16( f * (1 + sa) ), token-major (== NHWC)
__global__ void k_gate(const float* __restrict__ f, const float* __restrict__ sa,
                       u16* __restrict__ fg) {
  int i = blockIdx.x * 256 + threadIdx.x;
  if (i >= 4 * 784 * 256) return;
  fg[i] = f2bf(f[i] * (1.0f + sa[i >> 8]));
}

// g = bf16( BN_pbn( gelu( xcat + res ) ) ), NHWC [4,784,512]
__global__ void k_gelu_bn(const float* __restrict__ a, const float* __restrict__ b2,
                          const float* __restrict__ g, const float* __restrict__ bb,
                          const float* __restrict__ m, const float* __restrict__ v,
                          u16* __restrict__ out) {
  int i = blockIdx.x * 256 + threadIdx.x;
  if (i >= 4 * 784 * 512) return;
  int c = i & 511;
  float t = gelu_exact(a[i] + b2[i]);
  float sc = g[c] * rsqrtf(v[c] + 1e-5f);
  out[i] = f2bf(t * sc + (bb[c] - m[c] * sc));
}

// xc = p2out + x_attn   (both token-major [b,p,c], C=256)
__global__ void k_add(const float* __restrict__ a, const float* __restrict__ b,
                      float* __restrict__ o) {
  int i = blockIdx.x * 256 + threadIdx.x;
  if (i >= 4 * 784 * 256) return;
  o[i] = a[i] + b[i];
}

// depthwise 3x3 (pad 1) + bias + residual on NHWC; out tokens [b][p][c]
__global__ void k_dw_final(const float* __restrict__ xc,
                           const float* __restrict__ w,
                           const float* __restrict__ bias,
                           float* __restrict__ out) {
  int i = blockIdx.x * 256 + threadIdx.x;
  if (i >= 4 * 784 * 256) return;
  int c = i & 255;
  int pix = i >> 8;
  int b = pix / 784, p = pix % 784;
  int y = p / 28, x = p % 28;
  const float* base = xc + (size_t)b * 784 * 256 + c;
  float acc = base[(size_t)p * 256] + bias[c];
  for (int ky = 0; ky < 3; ++ky) {
    int iy = y - 1 + ky;
    if (iy < 0 || iy >= 28) continue;
    for (int kx = 0; kx < 3; ++kx) {
      int ix = x - 1 + kx;
      if (ix < 0 || ix >= 28) continue;
      acc += w[c * 9 + ky * 3 + kx] * base[(size_t)(iy * 28 + ix) * 256];
    }
  }
  out[i] = acc;
}

// ---------------------------------------------------------------------------
extern "C" void kernel_launch(void* const* d_in, const int* in_sizes, int n_in,
                              void* d_out, int out_size, void* d_ws, size_t ws_size,
                              hipStream_t stream) {
  (void)in_sizes; (void)n_in; (void)out_size; (void)ws_size;

  const float* X    = (const float*)d_in[0];   // [4,784,256]
  const float* REL  = (const float*)d_in[1];   // [8,784,784]
  const float* LN1G = (const float*)d_in[2];
  const float* LN1B = (const float*)d_in[3];
  const float* LN2G = (const float*)d_in[4];
  const float* LN2B = (const float*)d_in[5];
  const float* WQ   = (const float*)d_in[6];
  const float* WK   = (const float*)d_in[7];
  const float* WV   = (const float*)d_in[8];
  const float* WP   = (const float*)d_in[9];
  const float* BP   = (const float*)d_in[10];
  const float* SAW  = (const float*)d_in[11];
  const float* C1W  = (const float*)d_in[12];
  const float* C1B  = (const float*)d_in[13];
  const float* B1AW = (const float*)d_in[18];
  const float* B1BW = (const float*)d_in[23];
  const float* B2AW = (const float*)d_in[28];
  const float* B2BW = (const float*)d_in[33];
  const float* B3AW = (const float*)d_in[38];
  const float* B3BW = (const float*)d_in[43];
  const float* CATW = (const float*)d_in[48];
  const float* RESW = (const float*)d_in[53];
  const float* P2W  = (const float*)d_in[62];
  const float* P2B  = (const float*)d_in[63];
  const float* DWW  = (const float*)d_in[68];
  const float* DWB  = (const float*)d_in[69];
  float* OUT = (float*)d_out;

  // ---- workspace ----------------------------------------------------------
  size_t off = 0;
  auto take = [&](size_t bytes) -> void* {
    void* ptr = (char*)d_ws + off;
    off = (off + bytes + 255) & ~(size_t)255;
    return ptr;
  };
  const size_t TOK = 3136, HW = 784;

  u16* wqT = (u16*)take(65536 * 2);     // transposed weights [N][K]
  u16* wkT = (u16*)take(65536 * 2);
  u16* wvT = (u16*)take(65536 * 2);
  u16* wpT = (u16*)take(65536 * 2);
  u16* xn  = (u16*)take(TOK * 256 * 2);
  u16* qb  = (u16*)take(TOK * 256 * 2);
  u16* kb  = (u16*)take(TOK * 256 * 2);
  u16* vb  = (u16*)take(TOK * 256 * 2);
  u16* ob  = (u16*)take(TOK * 256 * 2);
  float* x_attn = (float*)take(TOK * 256 * 4);
  float* ftok   = (float*)take(TOK * 256 * 4);
  float* pool   = (float*)take(4 * 2 * HW * 4);
  float* samap  = (float*)take(4 * HW * 4);
  u16*   fg     = (u16*)take(TOK * 256 * 2);
  size_t arena = off;

  // phase A: attention scratch
  off = arena;
  float* S  = (float*)take((size_t)4 * 8 * 784 * 784 * 4);
  u16*   P  = (u16*)take((size_t)4 * 8 * 784 * 800 * 2);   // padded rows
  u16*   vt = (u16*)take((size_t)4 * 256 * 800 * 2);       // V transposed+pad

  // phase B: conv weights + activations (aliases phase A; stream-ordered)
  off = arena;
  u16* wc1  = (u16*)take((size_t)512 * 256 * 2);
  u16* wb1a = (u16*)take((size_t)512 * 512 * 9 * 2);
  u16* wb1b = (u16*)take((size_t)512 * 512 * 9 * 2);
  u16* wb2a = (u16*)take((size_t)512 * 512 * 25 * 2);
  u16* wb2b = (u16*)take((size_t)512 * 512 * 9 * 2);
  u16* wb3a = (u16*)take((size_t)512 * 512 * 49 * 2);
  u16* wb3b = (u16*)take((size_t)512 * 512 * 9 * 2);
  u16* wcat = (u16*)take((size_t)512 * 1536 * 9 * 2);
  u16* wres = (u16*)take((size_t)512 * 512 * 2);
  u16* wp2  = (u16*)take((size_t)256 * 512 * 2);
  u16*   F1b    = (u16*)take((size_t)4 * HW * 512 * 2);    // NHWC
  u16*   tmpb   = (u16*)take((size_t)4 * HW * 512 * 2);
  u16*   xcatin = (u16*)take((size_t)4 * HW * 1536 * 2);
  float* xcatf  = (float*)take((size_t)4 * HW * 512 * 4);
  float* resf   = (float*)take((size_t)4 * HW * 512 * 4);
  u16*   gbf    = (u16*)take((size_t)4 * HW * 512 * 2);
  float* p2of   = (float*)take((size_t)4 * HW * 256 * 4);
  float* xcf    = (float*)take((size_t)4 * HW * 256 * 4);

  auto packT = [&](const float* src, u16* dst, int K, int N) {
    k_pack_bf16_t<<<(K * N + 255) / 256, 256, 0, stream>>>(src, dst, K, N);
  };
  auto packW = [&](const float* src, u16* dst, int Co, int Ci, int KHW) {
    k_pack_convw<<<(Co * Ci * KHW + 255) / 256, 256, 0, stream>>>(src, dst, Co,
                                                                  Ci, KHW);
  };

  // ================= attention ================================
  packT(WQ, wqT, 256, 256); packT(WK, wkT, 256, 256);
  packT(WV, wvT, 256, 256); packT(WP, wpT, 256, 256);

  k_layernorm<<<392, 256, 0, stream>>>(X, LN1G, LN1B, xn, nullptr);

  // Q/K/V: M=3136 (49 tiles of 64), N=256 (16) -> 784 waves -> 98 blocks
  k_gemm_bf16<<<98, 256, 0, stream>>>(xn, 256, wqT, 256, 3136, 256, 256,
                                      nullptr, nullptr, nullptr, qb, 256);
  k_gemm_bf16<<<98, 256, 0, stream>>>(xn, 256, wkT, 256, 3136, 256, 256,
                                      nullptr, nullptr, nullptr, kb, 256);
  k_gemm_bf16<<<98, 256, 0, stream>>>(xn, 256, wvT, 256, 3136, 256, 256,
                                      nullptr, nullptr, nullptr, vb, 256);

  k_attn_score<<<(4 * 8 * 49 * 49 + 7) / 8, 256, 0, stream>>>(qb, kb, REL, S);
  k_softmax<<<4 * 8 * 784, 256, 0, stream>>>(S, P);
  k_vt<<<(4 * 256 * 800 + 255) / 256, 256, 0, stream>>>(vb, vt);
  k_attn_pv<<<(4 * 8 * 49 * 2 + 7) / 8, 256, 0, stream>>>(P, vt, ob);

  k_gemm_bf16<<<98, 256, 0, stream>>>(ob, 256, wpT, 256, 3136, 256, 256,
                                      BP, X, x_attn, nullptr, 256);

  // ================= conv MLP =================================
  k_layernorm<<<392, 256, 0, stream>>>(x_attn, LN2G, LN2B, nullptr, ftok);
  k_sa_pool<<<(4 * 784 + 255) / 256, 256, 0, stream>>>(ftok, pool);
  k_sa_conv<<<(4 * 784 + 255) / 256, 256, 0, stream>>>(pool, SAW, samap);
  k_gate<<<(4 * 784 * 256 + 255) / 256, 256, 0, stream>>>(ftok, samap, fg);

  packW(C1W, wc1, 512, 256, 1);
  packW(B1AW, wb1a, 512, 512, 9);
  packW(B1BW, wb1b, 512, 512, 9);
  packW(B2AW, wb2a, 512, 512, 25);
  packW(B2BW, wb2b, 512, 512, 9);
  packW(B3AW, wb3a, 512, 512, 49);
  packW(B3BW, wb3b, 512, 512, 9);
  packW(CATW, wcat, 512, 1536, 9);
  packW(RESW, wres, 512, 512, 1);
  packW(P2W, wp2, 256, 512, 1);

  // conv waves: 4 * (Co/64) * 49 ; 8 waves/block
  // Co=512 -> 1568 waves -> 196 blocks; Co=256 -> 784 waves -> 98 blocks
  // c1: 1x1, bias, gelu, BN
  k_conv_wmma<<<196, 256, 0, stream>>>(fg, 256, wc1, 1, 1, 0, 1, 512,
      C1B, 1, (const float*)d_in[14], (const float*)d_in[15],
      (const float*)d_in[16], (const float*)d_in[17], nullptr, F1b, 512, 0);
  // branch 1: 3x3 pad1 -> 3x3 pad3 dil3
  k_conv_wmma<<<196, 256, 0, stream>>>(F1b, 512, wb1a, 3, 3, 1, 1, 512,
      nullptr, 0, (const float*)d_in[19], (const float*)d_in[20],
      (const float*)d_in[21], (const float*)d_in[22], nullptr, tmpb, 512, 0);
  k_conv_wmma<<<196, 256, 0, stream>>>(tmpb, 512, wb1b, 3, 3, 3, 3, 512,
      nullptr, 0, (const float*)d_in[24], (const float*)d_in[25],
      (const float*)d_in[26], (const float*)d_in[27], nullptr, xcatin, 1536, 0);
  // branch 2: 5x5 pad2 -> 3x3 pad5 dil5
  k_conv_wmma<<<196, 256, 0, stream>>>(F1b, 512, wb2a, 5, 5, 2, 1, 512,
      nullptr, 0, (const float*)d_in[29], (const float*)d_in[30],
      (const float*)d_in[31], (const float*)d_in[32], nullptr, tmpb, 512, 0);
  k_conv_wmma<<<196, 256, 0, stream>>>(tmpb, 512, wb2b, 3, 3, 5, 5, 512,
      nullptr, 0, (const float*)d_in[34], (const float*)d_in[35],
      (const float*)d_in[36], (const float*)d_in[37], nullptr, xcatin, 1536, 512);
  // branch 3: 7x7 pad3 -> 3x3 pad7 dil7
  k_conv_wmma<<<196, 256, 0, stream>>>(F1b, 512, wb3a, 7, 7, 3, 1, 512,
      nullptr, 0, (const float*)d_in[39], (const float*)d_in[40],
      (const float*)d_in[41], (const float*)d_in[42], nullptr, tmpb, 512, 0);
  k_conv_wmma<<<196, 256, 0, stream>>>(tmpb, 512, wb3b, 3, 3, 7, 7, 512,
      nullptr, 0, (const float*)d_in[44], (const float*)d_in[45],
      (const float*)d_in[46], (const float*)d_in[47], nullptr, xcatin, 1536, 1024);
  // cat conv: Ci=1536, 3x3 pad1, BN -> f32
  k_conv_wmma<<<196, 256, 0, stream>>>(xcatin, 1536, wcat, 3, 3, 1, 1, 512,
      nullptr, 0, (const float*)d_in[49], (const float*)d_in[50],
      (const float*)d_in[51], (const float*)d_in[52], xcatf, nullptr, 512, 0);
  // res conv: 1x1, BN -> f32
  k_conv_wmma<<<196, 256, 0, stream>>>(F1b, 512, wres, 1, 1, 0, 1, 512,
      nullptr, 0, (const float*)d_in[54], (const float*)d_in[55],
      (const float*)d_in[56], (const float*)d_in[57], resf, nullptr, 512, 0);
  // g = BN_pbn(gelu(xcat + res))
  k_gelu_bn<<<(4 * 784 * 512 + 255) / 256, 256, 0, stream>>>(
      xcatf, resf, (const float*)d_in[58], (const float*)d_in[59],
      (const float*)d_in[60], (const float*)d_in[61], gbf);
  // p2: 1x1 512->256, bias, BN -> f32
  k_conv_wmma<<<98, 256, 0, stream>>>(gbf, 512, wp2, 1, 1, 0, 1, 256,
      P2B, 0, (const float*)d_in[64], (const float*)d_in[65],
      (const float*)d_in[66], (const float*)d_in[67], p2of, nullptr, 256, 0);
  // xc = p2out + x_attn (same token-major layout)
  k_add<<<(4 * 784 * 256 + 255) / 256, 256, 0, stream>>>(p2of, x_attn, xcf);
  // depthwise 3x3 + bias + residual -> tokens out
  k_dw_final<<<(4 * 784 * 256 + 255) / 256, 256, 0, stream>>>(xcf, DWW, DWB, OUT);
}
